// CrossAttention_24979529793594
// MI455X (gfx1250) — compile-verified, matrix operand order checked by reading, not run on hardware
//
#include <hip/hip_runtime.h>
#include <hip/hip_bf16.h>

#define N_NODES  50000
#define N_EDGES  800000
#define DIM      128
#define N_HEADS  8
#define HEAD_DIM 16
#define SCALE    0.25f   // HEAD_DIM^-0.5
#define LDS_STRIDE 136   // 128 + 8 halves pad: lanes hit distinct 4-bank groups
#define KV_ROWS  64      // edges per block in the kv kernel (4 row tiles/wave)

typedef _Float16 v16h __attribute__((ext_vector_type(16)));
typedef _Float16 v8h  __attribute__((ext_vector_type(8)));
typedef float    v8f  __attribute__((ext_vector_type(8)));

__device__ __forceinline__ v16h cat8(v8h a, v8h b) {
    return __builtin_shufflevector(a, b, 0,1,2,3,4,5,6,7,8,9,10,11,12,13,14,15);
}

__device__ __forceinline__ v8h cvt8(float4 x0, float4 x1) {
    v8h r;
    r[0] = (_Float16)x0.x; r[1] = (_Float16)x0.y; r[2] = (_Float16)x0.z; r[3] = (_Float16)x0.w;
    r[4] = (_Float16)x1.x; r[5] = (_Float16)x1.y; r[6] = (_Float16)x1.z; r[7] = (_Float16)x1.w;
    return r;
}

// Cooperative fp32->fp16 staging of ROWS x 128 A rows into LDS.
template <int ROWS>
__device__ __forceinline__ void stage_tile(const float* __restrict__ A, int row0,
                                           _Float16* As) {
#pragma unroll
    for (int i = 0; i < ROWS / 16; ++i) {
        int ch = threadIdx.x + 256 * i;      // chunk = 8 floats
        int r  = ch >> 4;
        int c0 = (ch & 15) * 8;
        const float4* ap =
            reinterpret_cast<const float4*>(A + (size_t)(row0 + r) * DIM + c0);
        float4 x0 = ap[0], x1 = ap[1];
        *reinterpret_cast<v8h*>(&As[r * LDS_STRIDE + c0]) = cvt8(x0, x1);
    }
}

// A fragment from LDS: halves 0..7 <-> K=kb+8hh+i ; 8..15 <-> K=kb+16+8hh+i
__device__ __forceinline__ v16h frag_a(const _Float16* As, int row, int kb, int hh) {
    const v8h* p0 = reinterpret_cast<const v8h*>(&As[row * LDS_STRIDE + kb + 8 * hh]);
    const v8h* p1 = reinterpret_cast<const v8h*>(&As[row * LDS_STRIDE + kb + 16 + 8 * hh]);
    return cat8(*p0, *p1);
}

// B fragment straight from pre-converted f16 weights (row n of W)
__device__ __forceinline__ v16h frag_b(const _Float16* __restrict__ wrow, int kb, int hh) {
    const v8h* p0 = reinterpret_cast<const v8h*>(wrow + kb + 8 * hh);
    const v8h* p1 = reinterpret_cast<const v8h*>(wrow + kb + 16 + 8 * hh);
    return cat8(*p0, *p1);
}

__device__ __forceinline__ void atomicMaxF32(float* addr, float val) {
    if (val >= 0.0f) atomicMax(reinterpret_cast<int*>(addr), __float_as_int(val));
    else             atomicMin(reinterpret_cast<unsigned int*>(addr), __float_as_uint(val));
}

// ---------------- K-1: convert weights to f16 (once per launch) ------------
__global__ void ga_cvt_w(const float* __restrict__ Wq, const float* __restrict__ Wkv,
                         const float* __restrict__ Wh, _Float16* __restrict__ wq16,
                         _Float16* __restrict__ wkv16, _Float16* __restrict__ wh16) {
    int i = blockIdx.x * 256 + threadIdx.x;
    if (i < DIM * DIM) {
        wq16[i] = (_Float16)Wq[i];
        wh16[i] = (_Float16)Wh[i];
    }
    if (i < 2 * DIM * DIM) wkv16[i] = (_Float16)Wkv[i];
}

// ---------------- K0: init scratch -----------------------------------------
__global__ void ga_init(float* __restrict__ segmax, float* __restrict__ segsum,
                        float* __restrict__ agg) {
    int i = blockIdx.x * 256 + threadIdx.x;
    if (i < N_NODES * DIM) agg[i] = 0.0f;
    if (i < N_NODES * N_HEADS) { segmax[i] = -3.4e38f; segsum[i] = 0.0f; }
}

// ---------------- K1/K5: out[M x 128] = A[M x 128] @ W16^T + bias ----------
__global__ void ga_gemm128(const float* __restrict__ A, const _Float16* __restrict__ W16,
                           const float* __restrict__ bias, float* __restrict__ out) {
    __shared__ __align__(16) _Float16 As[16 * LDS_STRIDE];
    int row0 = blockIdx.x * 16;
    stage_tile<16>(A, row0, As);
    __syncthreads();

    int lane = threadIdx.x & 31;
    int wid  = threadIdx.x >> 5;
    int nn   = lane & 15;
    int hh   = lane >> 4;
    const _Float16* wrow = W16 + (size_t)(wid * 16 + nn) * DIM;

    v8f c = {};
#pragma unroll
    for (int kb = 0; kb < DIM; kb += 32) {
        v16h a = frag_a(As, nn, kb, hh);
        v16h b = frag_b(wrow, kb, hh);
        c = __builtin_amdgcn_wmma_f32_16x16x32_f16(false, a, false, b,
                                                   (short)0, c, false, false);
    }
    int   col = wid * 16 + nn;
    float bn  = bias[col];
#pragma unroll
    for (int r = 0; r < 8; ++r) {
        int m = row0 + r + 8 * hh;
        out[(size_t)m * DIM + col] = c[r] + bn;
    }
}

// ---------------- K2: fused kv GEMM + f16 v store + attention scores -------
// 64 edges/block: wave `wid` owns head `wid` (k cols 16w.., v cols 128+16w..),
// accumulating 4 row tiles so each B fragment is reused 4x.
__global__ void ga_kv_score(const float* __restrict__ e, const _Float16* __restrict__ wkv16,
                            const float* __restrict__ bkv, const int* __restrict__ dst,
                            const float* __restrict__ q, _Float16* __restrict__ vbuf16,
                            float* __restrict__ attn, float* __restrict__ segmax) {
    __shared__ __align__(16) _Float16 As[KV_ROWS * LDS_STRIDE];
    int row0 = blockIdx.x * KV_ROWS;
    stage_tile<KV_ROWS>(e, row0, As);
    __syncthreads();

    int lane = threadIdx.x & 31;
    int wid  = threadIdx.x >> 5;          // head index
    int nn   = lane & 15;
    int hh   = lane >> 4;
    const _Float16* wkrow = wkv16 + (size_t)(wid * 16 + nn) * DIM;        // k half
    const _Float16* wvrow = wkv16 + (size_t)(DIM + wid * 16 + nn) * DIM;  // v half

    v8f ck[4] = {}, cv[4] = {};
#pragma unroll
    for (int kb = 0; kb < DIM; kb += 32) {
        v16h bk = frag_b(wkrow, kb, hh);
        v16h bv = frag_b(wvrow, kb, hh);
#pragma unroll
        for (int rt = 0; rt < 4; ++rt) {
            v16h a = frag_a(As, rt * 16 + nn, kb, hh);
            ck[rt] = __builtin_amdgcn_wmma_f32_16x16x32_f16(false, a, false, bk,
                                                            (short)0, ck[rt], false, false);
            cv[rt] = __builtin_amdgcn_wmma_f32_16x16x32_f16(false, a, false, bv,
                                                            (short)0, cv[rt], false, false);
        }
    }

    int   col = wid * 16 + nn;
    float bkn = bkv[col];
    float bvn = bkv[DIM + col];
#pragma unroll
    for (int rt = 0; rt < 4; ++rt) {
#pragma unroll
        for (int r = 0; r < 8; ++r) {
            int m = row0 + rt * 16 + r + 8 * hh;          // edge index for this C row
            float kval = ck[rt][r] + bkn;
            vbuf16[(size_t)m * DIM + col] = (_Float16)(cv[rt][r] + bvn);

            int   d    = dst[m];
            float prod = kval * q[(size_t)d * DIM + col]; // q gather: L2-resident
            prod += __shfl_xor(prod, 1, 16);
            prod += __shfl_xor(prod, 2, 16);
            prod += __shfl_xor(prod, 4, 16);
            prod += __shfl_xor(prod, 8, 16);
            if (nn == 0) {
                float s = prod * SCALE;
                attn[(size_t)m * N_HEADS + wid] = s;
                atomicMaxF32(&segmax[(size_t)d * N_HEADS + wid], s);
            }
        }
    }
}

// ---------------- K3: p = exp(s - max); seg_sum += p ------------------------
__global__ void ga_expsum(const int* __restrict__ dst, float* __restrict__ attn,
                          const float* __restrict__ segmax, float* __restrict__ segsum) {
    int idx = blockIdx.x * 256 + threadIdx.x;
    if (idx >= N_EDGES * N_HEADS) return;
    int ed = idx >> 3, hd = idx & 7;
    int d  = dst[ed];
    float p = __expf(attn[idx] - segmax[(size_t)d * N_HEADS + hd]);
    attn[idx] = p;
    unsafeAtomicAdd(&segsum[(size_t)d * N_HEADS + hd], p);
}

// ---------------- K4: agg[dst] += (p / sum) * v ----------------------------
__global__ void ga_aggregate(const int* __restrict__ dst, const float* __restrict__ attn,
                             const float* __restrict__ segsum,
                             const _Float16* __restrict__ vbuf16,
                             float* __restrict__ agg) {
    int t  = threadIdx.x;
    int ed = blockIdx.x * 2 + (t >> 7);
    int f  = t & 127;
    int hd = f >> 4;
    int d  = dst[ed];
    float w = attn[(size_t)ed * N_HEADS + hd] / segsum[(size_t)d * N_HEADS + hd];
    float v = (float)vbuf16[(size_t)ed * DIM + f];
    unsafeAtomicAdd(&agg[(size_t)d * DIM + f], w * v);
}

extern "C" void kernel_launch(void* const* d_in, const int* in_sizes, int n_in,
                              void* d_out, int out_size, void* d_ws, size_t ws_size,
                              hipStream_t stream) {
    const float* h_in = (const float*)d_in[0];
    const float* e_in = (const float*)d_in[1];
    const int*   dst  = (const int*)  d_in[2];
    const float* Wq   = (const float*)d_in[3];
    const float* bq   = (const float*)d_in[4];
    const float* Wkv  = (const float*)d_in[5];
    const float* bkv  = (const float*)d_in[6];
    const float* Wh   = (const float*)d_in[7];
    const float* bh   = (const float*)d_in[8];
    float* out = (float*)d_out;

    float* ws     = (float*)d_ws;
    float* q      = ws;                                  //  6,400,000 f32
    float* attn   = q      + (size_t)N_NODES * DIM;      //  6,400,000 f32
    float* segmax = attn   + (size_t)N_EDGES * N_HEADS;  //    400,000 f32
    float* segsum = segmax + (size_t)N_NODES * N_HEADS;  //    400,000 f32
    float* agg    = segsum + (size_t)N_NODES * N_HEADS;  //  6,400,000 f32
    _Float16* vbuf16 = (_Float16*)(agg + (size_t)N_NODES * DIM); // 102,400,000 f16 (~205MB)
    _Float16* wq16   = vbuf16 + (size_t)N_EDGES * DIM;
    _Float16* wkv16  = wq16  + DIM * DIM;                // f16 weights (~128KB)
    _Float16* wh16   = wkv16 + 2 * DIM * DIM;

    ga_cvt_w<<<(2 * DIM * DIM + 255) / 256, 256, 0, stream>>>(Wq, Wkv, Wh,
                                                              wq16, wkv16, wh16);
    ga_init<<<(N_NODES * DIM + 255) / 256, 256, 0, stream>>>(segmax, segsum, agg);
    ga_gemm128<<<N_NODES / 16, 256, 0, stream>>>(h_in, wq16, bq, q);
    ga_kv_score<<<N_EDGES / KV_ROWS, 256, 0, stream>>>(e_in, wkv16, bkv, dst, q,
                                                       vbuf16, attn, segmax);
    ga_expsum<<<(N_EDGES * N_HEADS + 255) / 256, 256, 0, stream>>>(dst, attn, segmax, segsum);
    ga_aggregate<<<N_EDGES / 2, 256, 0, stream>>>(dst, attn, segsum, vbuf16, agg);
    ga_gemm128<<<N_NODES / 16, 256, 0, stream>>>(agg, wh16, bh, out);
}
